// loss_3839700763115
// MI455X (gfx1250) — matrix operand order repaired and checked by reading, not compiled, not run on hardware
//
#include <hip/hip_runtime.h>
#include <math.h>

// ---------------------------------------------------------------------------
// PINN loss: 4-layer tanh MLP + nested-JVP second derivative, on gfx1250.
// Hidden-layer matrix products (value / 1st tangent / 2nd tangent, shared W)
// run on v_wmma_f32_16x16x32_f16; derivative algebra in f32 VALU with
// hardware v_tanh_f32.
// ---------------------------------------------------------------------------

#define NPTS    1000000
#define NN      32
#define TILE_M  16
#define NTILES  (NPTS / TILE_M)   // 62500 exactly
#define BLOCK   256
#define WAVES   (BLOCK / 32)
#define GRID    1024

typedef __attribute__((ext_vector_type(16))) _Float16 v16h;
typedef __attribute__((ext_vector_type(8)))  float    v8f;

__device__ __forceinline__ float fast_tanh(float v) {
#if __has_builtin(__builtin_amdgcn_tanhf)
  return __builtin_amdgcn_tanhf(v);        // v_tanh_f32 (gfx1250 TRANS op)
#else
  // branch-free fallback: 1 - 2/(e^{2x}+1); saturates to +/-1 correctly
  float e = __expf(2.0f * v);
  return 1.0f - 2.0f / (e + 1.0f);
#endif
}

__device__ __forceinline__ void lds_fence() {
  // DS pipe is in-order within a wave; this pins compiler ordering across the
  // f16/f32 aliased staging phases and drains outstanding DS ops.
  asm volatile("s_wait_dscnt 0" ::: "memory");
  __builtin_amdgcn_wave_barrier();
}

// A-matrix 16x32 f16 fragment (CDNA5 ISA 7.12.2):
//   lanes 0-15 : row = lane,    VGPR0-3 K=0..7,  VGPR4-7 K=16..23
//   lanes16-31 : row = lane-16, VGPR0-3 K=8..15, VGPR4-7 K=24..31
__device__ __forceinline__ v16h load_a_frag(const _Float16* buf, int m, int khalf) {
  union { uint4 u[2]; v16h v; } U;
  const _Float16* p = buf + m * NN + khalf * 8;
  U.u[0] = *(const uint4*)(p);          // K = khalf*8 .. khalf*8+7
  U.u[1] = *(const uint4*)(p + 16);     // K = 16+khalf*8 .. +7
  return U.v;
}

// B-matrix 32x16 f16 fragment from transposed weights wt[n*NN + k]:
//   lanes 0-15 : col = lane,    K=0..7,  K=16..23
//   lanes16-31 : col = lane-16, K=8..15, K=24..31
__device__ __forceinline__ v16h load_b_frag(const _Float16* wt, int n, int khalf) {
  union { uint4 u[2]; v16h v; } U;
  const _Float16* p = wt + n * NN + khalf * 8;
  U.u[0] = *(const uint4*)(p);
  U.u[1] = *(const uint4*)(p + 16);
  return U.v;
}

// ---------------------------------------------------------------------------
// Boundary term u(pi)^2 + u(0)^2 — one thread; also INITIALIZES out[0].
// ---------------------------------------------------------------------------
__global__ void pinn_boundary_kernel(const float* __restrict__ W1, const float* __restrict__ b1,
                                     const float* __restrict__ W2, const float* __restrict__ b2,
                                     const float* __restrict__ W3, const float* __restrict__ b3,
                                     const float* __restrict__ W4, const float* __restrict__ b4,
                                     float* __restrict__ out) {
  if (threadIdx.x != 0 || blockIdx.x != 0) return;
  float acc = 0.0f;
  const float xb[2] = {3.14159265358979323846f, 0.0f};
  for (int q = 0; q < 2; ++q) {
    float h[NN], g[NN];
    for (int j = 0; j < NN; ++j) h[j] = fast_tanh(xb[q] * W1[j] + b1[j]);
    for (int n = 0; n < NN; ++n) {
      float z = b2[n];
      for (int k = 0; k < NN; ++k) z += h[k] * W2[k * NN + n];
      g[n] = fast_tanh(z);
    }
    for (int n = 0; n < NN; ++n) {
      float z = b3[n];
      for (int k = 0; k < NN; ++k) z += g[k] * W3[k * NN + n];
      h[n] = fast_tanh(z);
    }
    float z = b4[0];
    for (int k = 0; k < NN; ++k) z += h[k] * W4[k];
    float u = fast_tanh(z);
    acc += u * u;
  }
  out[0] = acc;   // initialize accumulator with boundary term
}

// ---------------------------------------------------------------------------
// Main kernel: one 16-point tile per wave-iteration.
// ---------------------------------------------------------------------------
__global__ __launch_bounds__(BLOCK) void pinn_wmma_kernel(
    const float* __restrict__ x,
    const float* __restrict__ W1, const float* __restrict__ b1,
    const float* __restrict__ W2, const float* __restrict__ b2,
    const float* __restrict__ W3, const float* __restrict__ b3,
    const float* __restrict__ W4, const float* __restrict__ b4,
    float* __restrict__ out)
{
  __shared__ _Float16 sW2t[NN * NN];               // transposed [n][k], f16
  __shared__ _Float16 sW3t[NN * NN];
  // Per-wave staging: 3 matrices x 16 rows x 32 cols, reused as f16 then f32.
  __shared__ float    sStage[WAVES][3 * TILE_M * NN];   // 6 KB per wave

  // ---- stage hidden weights (transpose + f32->f16) ----
  for (int idx = threadIdx.x; idx < NN * NN; idx += BLOCK) {
    int k = idx / NN, n = idx % NN;
    sW2t[n * NN + k] = (_Float16)W2[idx];
    sW3t[n * NN + k] = (_Float16)W3[idx];
  }
  __syncthreads();

  const int lane  = threadIdx.x & 31;
  const int wv    = threadIdx.x >> 5;
  const int nlo   = lane & 15;        // row (point) / col (feature) sub-index
  const int khalf = lane >> 4;        // which K-half this lane covers
  const int j0    = khalf * 16;       // this lane's 16-feature slice

  // Loop-invariant B fragments: W2/W3, low and high 16 output columns.
  v16h B2f[2], B3f[2];
#pragma unroll
  for (int t = 0; t < 2; ++t) {
    B2f[t] = load_b_frag(sW2t, nlo + 16 * t, khalf);
    B3f[t] = load_b_frag(sW3t, nlo + 16 * t, khalf);
  }

  // Bias fragments for the value path: C element (r) maps to col nlo+16t,
  // so the exact bias fragment is a splat of b[col].
  v8f cb2[2], cb3[2];
#pragma unroll
  for (int t = 0; t < 2; ++t) {
    float v2 = b2[nlo + 16 * t], v3 = b3[nlo + 16 * t];
#pragma unroll
    for (int r = 0; r < 8; ++r) { cb2[t][r] = v2; cb3[t][r] = v3; }
  }
  const float b4s = b4[0];

  // Per-lane register copies of the tiny vectors (constant-indexed below).
  float w1r[16], b1r[16], w4r[16];
#pragma unroll
  for (int i = 0; i < 16; ++i) {
    w1r[i] = W1[j0 + i]; b1r[i] = b1[j0 + i]; w4r[i] = W4[j0 + i];
  }

  float*    fb = sStage[wv];               // f32 view (final staging)
  _Float16* hb = (_Float16*)sStage[wv];    // f16 view (WMMA A staging)

  const int waveId = blockIdx.x * WAVES + wv;
  const int nWaves = gridDim.x * WAVES;

  float acc = 0.0f;
  const v8f czero = {};

  for (int tile = waveId; tile < NTILES; tile += nWaves) {
    const int base = tile * TILE_M;
    const float xi = x[base + nlo];                      // lanes 16-31 dup row
    __builtin_prefetch(x + base + TILE_M * nWaves, 0, 1); // -> global_prefetch_b8

    // ---- layer 1 (1->32), scalar; JVP seed dx=1: z'=W1[j], z''=0 ----
    // lane (nlo,khalf) produces row nlo, features j0..j0+15 (contiguous).
    {
      union { _Float16 h[16]; uint4 u[2]; } pa, pp, pq;
#pragma unroll
      for (int i = 0; i < 16; ++i) {
        float w   = w1r[i];
        float a   = fast_tanh(xi * w + b1r[i]);
        float s   = 1.0f - a * a;
        float ap  = s * w;
        float app = -2.0f * a * ap * w;
        pa.h[i] = (_Float16)a; pp.h[i] = (_Float16)ap; pq.h[i] = (_Float16)app;
      }
      uint4* d0 = (uint4*)&hb[0 * 512 + nlo * NN + j0];
      uint4* d1 = (uint4*)&hb[1 * 512 + nlo * NN + j0];
      uint4* d2 = (uint4*)&hb[2 * 512 + nlo * NN + j0];
      d0[0] = pa.u[0]; d0[1] = pa.u[1];
      d1[0] = pp.u[0]; d1[1] = pp.u[1];
      d2[0] = pq.u[0]; d2[1] = pq.u[1];
    }
    lds_fence();

    // ================= layer 2 (32->32) on WMMA =================
    v16h A0 = load_a_frag(hb + 0 * 512, nlo, khalf);
    v16h A1 = load_a_frag(hb + 1 * 512, nlo, khalf);
    v16h A2 = load_a_frag(hb + 2 * 512, nlo, khalf);
    lds_fence();

    v8f zH[2], zP[2], zQ[2];
#pragma unroll
    for (int t = 0; t < 2; ++t) {
      zH[t] = __builtin_amdgcn_wmma_f32_16x16x32_f16(false, A0, false, B2f[t], (short)0, cb2[t], false, false);
      zP[t] = __builtin_amdgcn_wmma_f32_16x16x32_f16(false, A1, false, B2f[t], (short)0, czero,  false, false);
      zQ[t] = __builtin_amdgcn_wmma_f32_16x16x32_f16(false, A2, false, B2f[t], (short)0, czero,  false, false);
    }
    // nonlinearity + derivative chain, restage as f16
#pragma unroll
    for (int t = 0; t < 2; ++t) {
#pragma unroll
      for (int r = 0; r < 8; ++r) {
        float a   = fast_tanh(zH[t][r]);
        float s   = 1.0f - a * a;
        float zp  = zP[t][r];
        float ap  = s * zp;
        float app = s * zQ[t][r] - 2.0f * a * ap * zp;
        int m   = r + khalf * 8;                // C layout: row = vgpr + 8*(lane>=16)
        int off = m * NN + (nlo + 16 * t);      //           col = lane&15 (+16 per tile)
        hb[0 * 512 + off] = (_Float16)a;
        hb[1 * 512 + off] = (_Float16)ap;
        hb[2 * 512 + off] = (_Float16)app;
      }
    }
    lds_fence();

    // ================= layer 3 (32->32) on WMMA =================
    A0 = load_a_frag(hb + 0 * 512, nlo, khalf);
    A1 = load_a_frag(hb + 1 * 512, nlo, khalf);
    A2 = load_a_frag(hb + 2 * 512, nlo, khalf);
    lds_fence();

#pragma unroll
    for (int t = 0; t < 2; ++t) {
      zH[t] = __builtin_amdgcn_wmma_f32_16x16x32_f16(false, A0, false, B3f[t], (short)0, cb3[t], false, false);
      zP[t] = __builtin_amdgcn_wmma_f32_16x16x32_f16(false, A1, false, B3f[t], (short)0, czero,  false, false);
      zQ[t] = __builtin_amdgcn_wmma_f32_16x16x32_f16(false, A2, false, B3f[t], (short)0, czero,  false, false);
    }
    // nonlinearity, stage f32 for the final reduction
#pragma unroll
    for (int t = 0; t < 2; ++t) {
#pragma unroll
      for (int r = 0; r < 8; ++r) {
        float a   = fast_tanh(zH[t][r]);
        float s   = 1.0f - a * a;
        float zp  = zP[t][r];
        float ap  = s * zp;
        float app = s * zQ[t][r] - 2.0f * a * ap * zp;
        int m   = r + khalf * 8;
        int off = m * NN + (nlo + 16 * t);
        fb[0 * 512 + off] = a;
        fb[1 * 512 + off] = ap;
        fb[2 * 512 + off] = app;
      }
    }
    lds_fence();

    // ---- layer 4 (32->1), per-point dot; each lane-half sums 16 terms ----
    float s0p = 0.0f, s1p = 0.0f, s2p = 0.0f;
    {
      const float4* r0 = (const float4*)&fb[0 * 512 + nlo * NN + j0];
      const float4* r1 = (const float4*)&fb[1 * 512 + nlo * NN + j0];
      const float4* r2 = (const float4*)&fb[2 * 512 + nlo * NN + j0];
#pragma unroll
      for (int q = 0; q < 4; ++q) {
        float4 g0 = r0[q], g1 = r1[q], g2 = r2[q];
        s0p += g0.x * w4r[4*q+0] + g0.y * w4r[4*q+1] + g0.z * w4r[4*q+2] + g0.w * w4r[4*q+3];
        s1p += g1.x * w4r[4*q+0] + g1.y * w4r[4*q+1] + g1.z * w4r[4*q+2] + g1.w * w4r[4*q+3];
        s2p += g2.x * w4r[4*q+0] + g2.y * w4r[4*q+1] + g2.z * w4r[4*q+2] + g2.w * w4r[4*q+3];
      }
    }
    float s0 = s0p + __shfl_xor(s0p, 16, 32) + b4s;
    float s1 = s1p + __shfl_xor(s1p, 16, 32);
    float s2 = s2p + __shfl_xor(s2p, 16, 32);

    float u   = fast_tanh(s0);
    float s   = 1.0f - u * u;
    float up  = s * s1;
    float upp = s * s2 - 2.0f * u * up * s1;
    float r   = upp - __sinf(xi);
    acc += (khalf == 0) ? r * r : 0.0f;      // count each point once
    lds_fence();
  }

  // ---- wave reduction, one atomic per wave, pre-scaled for the mean ----
#pragma unroll
  for (int off = 16; off > 0; off >>= 1) acc += __shfl_xor(acc, off, 32);
  if (lane == 0) atomicAdd(out, acc * (1.0f / (float)NPTS));
}

// ---------------------------------------------------------------------------
extern "C" void kernel_launch(void* const* d_in, const int* in_sizes, int n_in,
                              void* d_out, int out_size, void* d_ws, size_t ws_size,
                              hipStream_t stream) {
  (void)in_sizes; (void)n_in; (void)out_size; (void)d_ws; (void)ws_size;
  const float* x  = (const float*)d_in[0];
  const float* W1 = (const float*)d_in[1];
  const float* b1 = (const float*)d_in[2];
  const float* W2 = (const float*)d_in[3];
  const float* b2 = (const float*)d_in[4];
  const float* W3 = (const float*)d_in[5];
  const float* b3 = (const float*)d_in[6];
  const float* W4 = (const float*)d_in[7];
  const float* b4 = (const float*)d_in[8];
  float* out = (float*)d_out;

  // 1) initialize out[0] with the boundary term (also un-poisons d_out)
  pinn_boundary_kernel<<<1, 1, 0, stream>>>(W1, b1, W2, b2, W3, b3, W4, b4, out);
  // 2) PDE residual MSE accumulated on top
  pinn_wmma_kernel<<<GRID, BLOCK, 0, stream>>>(x, W1, b1, W2, b2, W3, b3, W4, b4, out);
}
